// MimiResidualVectorQuantizer_54322746359968
// MI455X (gfx1250) — matrix operand order repaired and checked by reading, not compiled, not run on hardware
//
#include <hip/hip_runtime.h>

// ---------------- problem constants ----------------
#define T_DIM 65536
#define D_DIM 512
#define C_DIM 256
#define K_DIM 2048
#define Q_DIM 8
#define RPB   128          // token rows per block (8 waves x 16 rows)
#define AST   (D_DIM + 8)  // padded LDS row stride (bank-conflict-free: +4 dwords/row)
#define GST   (C_DIM + 8)  // padded LDS row stride for xp/gather buffer

typedef __attribute__((ext_vector_type(16))) __bf16 v16bf;
typedef __attribute__((ext_vector_type(8)))  float  v8f;

union ABu { v16bf v; uint4 q[2]; };

__device__ __forceinline__ unsigned short f2bf(float f) {
  unsigned u = __float_as_uint(f);
  u += 0x7FFFu + ((u >> 16) & 1u);          // round-to-nearest-even
  return (unsigned short)(u >> 16);
}

__device__ __forceinline__ v8f vzero() {
  v8f z = {0.f, 0.f, 0.f, 0.f, 0.f, 0.f, 0.f, 0.f};
  return z;
}

__device__ __forceinline__ v8f wmma_bf16(const ABu& a, const ABu& b, v8f c) {
  return __builtin_amdgcn_wmma_f32_16x16x32_bf16(
      /*neg_a=*/false, a.v, /*neg_b=*/false, b.v,
      /*c_mod=*/(short)0, c, /*reuse_a=*/false, /*reuse_b=*/false);
}

__device__ __forceinline__ void load_a(ABu& a, const unsigned short* ap) {
  a.q[0] = *(const uint4*)(ap);
  a.q[1] = *(const uint4*)(ap + 8);    // contiguous 16 K-values per lane (A striping)
}
__device__ __forceinline__ void load_b(ABu& b, const unsigned short* bp) {
  b.q[0] = *(const uint4*)(bp);
  b.q[1] = *(const uint4*)(bp + 8);    // contiguous 16 K-values per lane (B striping)
}

// ---------------- init: res = x, recon = 0 ----------------
__global__ void rvq_init(const float4* __restrict__ x, float4* __restrict__ res,
                         float4* __restrict__ recon, int n4) {
  int i = blockIdx.x * blockDim.x + threadIdx.x;
  if (i < n4) {
    res[i] = x[i];
    float4 z; z.x = z.y = z.z = z.w = 0.f;
    recon[i] = z;
  }
}

// ---------------- convert all stage weights f32 -> bf16 ----------------
__global__ void rvq_convert(const float* __restrict__ w_in, const float* __restrict__ w_out,
                            const float* __restrict__ e,
                            unsigned short* __restrict__ w_in_bf,
                            unsigned short* __restrict__ w_out_bf,
                            unsigned short* __restrict__ e_bf) {
  size_t gid = (size_t)blockIdx.x * blockDim.x + threadIdx.x;
  const size_t n_in  = (size_t)Q_DIM * C_DIM * D_DIM;
  const size_t n_out = (size_t)Q_DIM * D_DIM * C_DIM;
  const size_t n_e   = (size_t)Q_DIM * K_DIM * C_DIM;
  if (gid < n_in) {
    w_in_bf[gid] = f2bf(w_in[gid]);
  } else if (gid < n_in + n_out) {
    size_t i = gid - n_in;
    w_out_bf[i] = f2bf(w_out[i]);
  } else if (gid < n_in + n_out + n_e) {
    size_t i = gid - n_in - n_out;
    e_bf[i] = f2bf(e[i]);
  }
}

// ---------------- |e_k|^2 for all stages ----------------
__global__ void rvq_esq(const float* __restrict__ e, float* __restrict__ esq) {
  int gid = blockIdx.x * blockDim.x + threadIdx.x;
  if (gid < Q_DIM * K_DIM) {
    const float4* row = (const float4*)(e + (size_t)gid * C_DIM);
    float s = 0.f;
#pragma unroll 4
    for (int i = 0; i < C_DIM / 4; ++i) {
      float4 v = row[i];
      s += v.x * v.x + v.y * v.y + v.z * v.z + v.w * v.w;
    }
    esq[gid] = s;
  }
}

// ---------------- fused RVQ stage: project, argmin, decode, update ----------------
__global__ __launch_bounds__(256) void rvq_stage(
    float* __restrict__ res,                       // [T,D] f32, updated in place
    const unsigned short* __restrict__ w_in_bf,    // [C,D] bf16
    const unsigned short* __restrict__ e_bf,       // [K,C] bf16
    const unsigned short* __restrict__ w_out_bf,   // [D,C] bf16
    const float* __restrict__ esq,                 // [K]   f32
    float* __restrict__ recon,                     // [T,D] f32, accumulated
    float* __restrict__ codes)                     // [T]   float-encoded indices
{
  extern __shared__ unsigned short smem[];
  unsigned short* Abuf = smem;                      // [RPB][AST] staged residual (bf16)
  unsigned short* Gbuf = smem + RPB * AST;          // [RPB][GST] xp, then gathered codes
  float*          Ebuf = (float*)(smem + RPB * AST + RPB * GST);  // [K] |e|^2

  const int lane  = threadIdx.x & 31;
  const int wave  = threadIdx.x >> 5;
  const int grp   = lane >> 4;                      // lane-half (CDNA5 16x16 striping)
  const int l15   = lane & 15;
  const int wrow0 = wave * 16;                      // each wave owns 16 rows
  const int trow0 = blockIdx.x * RPB;

  // ---- stage |e|^2 into LDS (block-cooperative, single barrier) ----
  for (int i = threadIdx.x; i < K_DIM; i += 256) Ebuf[i] = esq[i];

  // ---- stage this wave's residual rows (f32 -> bf16) into LDS, padded stride ----
  {
    const float* src = res + (size_t)(trow0 + wrow0) * D_DIM;
    for (int i = lane * 4; i < 16 * D_DIM; i += 32 * 4) {
      const int r = i >> 9;          // i / D_DIM
      const int c = i & (D_DIM - 1);
      float4 f = *(const float4*)(src + i);
      ushort4 h;
      h.x = f2bf(f.x); h.y = f2bf(f.y); h.z = f2bf(f.z); h.w = f2bf(f.w);
      *(ushort4*)(Abuf + (wrow0 + r) * AST + c) = h;
    }
  }
  __syncthreads();   // only for Ebuf visibility; A/G buffers stay wave-private

  // ---- phase 1: xp = res @ w_in^T, 2-way N blocking, track |xp|^2 ----
  float xsq[8];
#pragma unroll
  for (int j = 0; j < 8; ++j) xsq[j] = 0.f;

  const unsigned short* arow = Abuf + (wrow0 + l15) * AST;
  for (int nt = 0; nt < C_DIM / 16; nt += 2) {
    v8f acc0 = vzero(), acc1 = vzero();
    const unsigned short* brow0 = w_in_bf + (nt * 16 + l15) * D_DIM;
    const unsigned short* brow1 = brow0 + 16 * D_DIM;
#pragma unroll 4
    for (int kc = 0; kc < D_DIM / 32; ++kc) {
      ABu a, b0, b1;
      load_a(a, arow + kc * 32 + grp * 16);
      load_b(b0, brow0 + kc * 32 + grp * 16);
      load_b(b1, brow1 + kc * 32 + grp * 16);
      acc0 = wmma_bf16(a, b0, acc0);
      acc1 = wmma_bf16(a, b1, acc1);
    }
#pragma unroll
    for (int j = 0; j < 8; ++j) {
      float v0 = acc0[j], v1 = acc1[j];
      xsq[j] += v0 * v0 + v1 * v1;
      unsigned short* gp = Gbuf + (wrow0 + j + grp * 8) * GST + nt * 16 + l15;
      gp[0]  = f2bf(v0);
      gp[16] = f2bf(v1);
    }
  }
  // reduce |xp|^2 across the 16 lanes sharing each C/D-layout row
#pragma unroll
  for (int j = 0; j < 8; ++j) {
#pragma unroll
    for (int m = 1; m < 16; m <<= 1) xsq[j] += __shfl_xor(xsq[j], m, 16);
  }

  // ---- phase 2: dist = |xp|^2 - 2 xp.e + |e|^2, argmin over K, 2-way N blocking ----
  float minval[8];
  int   minidx[8];
#pragma unroll
  for (int j = 0; j < 8; ++j) { minval[j] = 3.4e38f; minidx[j] = 0; }

  const unsigned short* xrow = Gbuf + (wrow0 + l15) * GST;
  for (int nt = 0; nt < K_DIM / 16; nt += 2) {
    const int n0 = nt * 16 + l15;
    const int n1 = n0 + 16;
    const unsigned short* brow0 = e_bf + (size_t)n0 * C_DIM;
    const unsigned short* brow1 = e_bf + (size_t)n1 * C_DIM;
    if (nt + 2 < K_DIM / 16) __builtin_prefetch(brow0 + 32 * C_DIM, 0, 0);
    v8f acc0 = vzero(), acc1 = vzero();
#pragma unroll
    for (int kc = 0; kc < C_DIM / 32; ++kc) {
      ABu a, b0, b1;
      load_a(a, xrow + kc * 32 + grp * 16);
      load_b(b0, brow0 + kc * 32 + grp * 16);
      load_b(b1, brow1 + kc * 32 + grp * 16);
      acc0 = wmma_bf16(a, b0, acc0);
      acc1 = wmma_bf16(a, b1, acc1);
    }
    const float eq0 = Ebuf[n0];
    const float eq1 = Ebuf[n1];
#pragma unroll
    for (int j = 0; j < 8; ++j) {
      float d0 = xsq[j] - 2.f * acc0[j] + eq0;
      float d1 = xsq[j] - 2.f * acc1[j] + eq1;
      if (d0 < minval[j]) { minval[j] = d0; minidx[j] = n0; }
      if (d1 < minval[j]) { minval[j] = d1; minidx[j] = n1; }
    }
  }
  // width-16 butterfly argmin (ties -> lowest index, matching argmin)
#pragma unroll
  for (int j = 0; j < 8; ++j) {
#pragma unroll
    for (int m = 1; m < 16; m <<= 1) {
      float ov = __shfl_xor(minval[j], m, 16);
      int   oi = __shfl_xor(minidx[j], m, 16);
      if (ov < minval[j] || (ov == minval[j] && oi < minidx[j])) {
        minval[j] = ov; minidx[j] = oi;
      }
    }
  }
  if (l15 == 0) {
#pragma unroll
    for (int j = 0; j < 8; ++j)
      codes[trow0 + wrow0 + j + grp * 8] = (float)minidx[j];
  }

  // ---- gather winning codebook rows (bf16) into LDS (reuse xp buffer) ----
#pragma unroll
  for (int j = 0; j < 8; ++j) {
    const unsigned short* srcp = e_bf + (size_t)minidx[j] * C_DIM + l15 * 16;
    unsigned short* dstp = Gbuf + (wrow0 + j + grp * 8) * GST + l15 * 16;
    *(uint4*)(dstp)     = *(const uint4*)(srcp);
    *(uint4*)(dstp + 8) = *(const uint4*)(srcp + 8);
  }

  // ---- phase 3: q = e_sel @ w_out^T ; res -= q ; recon += q ; 2-way N blocking ----
  const unsigned short* grow = Gbuf + (wrow0 + l15) * GST;
  for (int nt = 0; nt < D_DIM / 16; nt += 2) {
    v8f acc0 = vzero(), acc1 = vzero();
    const unsigned short* brow0 = w_out_bf + (nt * 16 + l15) * C_DIM;
    const unsigned short* brow1 = brow0 + 16 * C_DIM;
#pragma unroll
    for (int kc = 0; kc < C_DIM / 32; ++kc) {
      ABu a, b0, b1;
      load_a(a, grow + kc * 32 + grp * 16);
      load_b(b0, brow0 + kc * 32 + grp * 16);
      load_b(b1, brow1 + kc * 32 + grp * 16);
      acc0 = wmma_bf16(a, b0, acc0);
      acc1 = wmma_bf16(a, b1, acc1);
    }
#pragma unroll
    for (int j = 0; j < 8; ++j) {
      size_t off = (size_t)(trow0 + wrow0 + j + grp * 8) * D_DIM + nt * 16 + l15;
      float q0 = acc0[j], q1 = acc1[j];
      res[off]        -= q0;
      recon[off]      += q0;
      res[off + 16]   -= q1;
      recon[off + 16] += q1;
    }
  }
}

// ---------------- host-side orchestration ----------------
extern "C" void kernel_launch(void* const* d_in, const int* in_sizes, int n_in,
                              void* d_out, int out_size, void* d_ws, size_t ws_size,
                              hipStream_t stream) {
  (void)in_sizes; (void)n_in; (void)out_size; (void)ws_size;

  const float* x     = (const float*)d_in[0];   // [T,D]
  const float* w_in  = (const float*)d_in[1];   // [Q,C,D]
  const float* w_out = (const float*)d_in[2];   // [Q,D,C]
  const float* e     = (const float*)d_in[3];   // [Q,K,C]

  float* recon = (float*)d_out;                               // [T,D]
  float* codes = (float*)d_out + (size_t)T_DIM * D_DIM;       // [Q,T] (float-encoded)

  // workspace layout
  char* ws = (char*)d_ws;
  const size_t RES_B  = (size_t)T_DIM * D_DIM * sizeof(float);
  const size_t WIN_B  = (size_t)Q_DIM * C_DIM * D_DIM * sizeof(unsigned short);
  const size_t WOUT_B = (size_t)Q_DIM * D_DIM * C_DIM * sizeof(unsigned short);
  const size_t E_B    = (size_t)Q_DIM * K_DIM * C_DIM * sizeof(unsigned short);
  float*          res_ws   = (float*)ws;
  unsigned short* w_in_bf  = (unsigned short*)(ws + RES_B);
  unsigned short* w_out_bf = (unsigned short*)(ws + RES_B + WIN_B);
  unsigned short* e_bf     = (unsigned short*)(ws + RES_B + WIN_B + WOUT_B);
  float*          esq_ws   = (float*)(ws + RES_B + WIN_B + WOUT_B + E_B);

  // 1) res = x, recon = 0
  {
    int n4 = T_DIM * D_DIM / 4;
    rvq_init<<<(n4 + 255) / 256, 256, 0, stream>>>(
        (const float4*)x, (float4*)res_ws, (float4*)recon, n4);
  }
  // 2) bf16 weight images for all stages
  {
    size_t total = (size_t)Q_DIM * C_DIM * D_DIM * 2 + (size_t)Q_DIM * K_DIM * C_DIM;
    rvq_convert<<<(unsigned)((total + 255) / 256), 256, 0, stream>>>(
        w_in, w_out, e, w_in_bf, w_out_bf, e_bf);
    rvq_esq<<<(Q_DIM * K_DIM + 255) / 256, 256, 0, stream>>>(e, esq_ws);
  }
  // 3) eight fused encode+decode stages (serialized via stream => residual chain)
  const size_t SMEM = (size_t)(RPB * AST + RPB * GST) * sizeof(unsigned short)
                    + (size_t)K_DIM * sizeof(float);   // 208,896 B (< 320 KB/WGP)
  for (int q = 0; q < Q_DIM; ++q) {
    rvq_stage<<<T_DIM / RPB, 256, SMEM, stream>>>(
        res_ws,
        w_in_bf  + (size_t)q * C_DIM * D_DIM,
        e_bf     + (size_t)q * K_DIM * C_DIM,
        w_out_bf + (size_t)q * D_DIM * C_DIM,
        esq_ws   + (size_t)q * K_DIM,
        recon,
        codes    + (size_t)q * T_DIM);
  }
}
